// SpatialAttention_81698867904705
// MI455X (gfx1250) — matrix operand order repaired and checked by reading, not compiled, not run on hardware
//
#include <hip/hip_runtime.h>

typedef __attribute__((ext_vector_type(8)))  _Float16 v8h;
typedef __attribute__((ext_vector_type(16))) _Float16 v16h;
typedef __attribute__((ext_vector_type(2)))  _Float16 v2h;
typedef __attribute__((ext_vector_type(8)))  float    v8f;

#define LN_EPS 1e-5f
constexpr int N  = 512;   // tokens
constexpr int F  = 128;   // features
constexpr int IPB = 8;    // i per block (= waves per block)

// ---------------------------------------------------------------------------
// Fragment loaders (CDNA5 WMMA 16x16x32 f16 layouts, wave32)
// ---------------------------------------------------------------------------
// A (16x32 f16): lane l holds row m=l&15; lanes 0-15: K {8g..} per table.
// Stored row-major f16[16][128]; per k-step two contiguous 8-half runs.
static __device__ __forceinline__ v16h cat8(v8h a, v8h b) {
  return __builtin_shufflevector(a, b, 0,1,2,3,4,5,6,7,8,9,10,11,12,13,14,15);
}

static __device__ __forceinline__ v16h load_afrag(const _Float16* A, int lane, int s) {
  int m = lane & 15, g = lane >> 4;
  const _Float16* p = A + m * 128 + s * 32 + g * 8;  // K = 32s + 8g + [0..7]
  v8h lo = *(const v8h*)p;                           // VGPRs 0-3
  v8h hi = *(const v8h*)(p + 16);                    // K = 32s + 16 + 8g + [0..7]
  return cat8(lo, hi);
}

// B stored frag-major in LDS: u32 index = ((t*4+s)*32 + lane)*8 + v
static __device__ __forceinline__ v16h load_bfrag(const unsigned int* B, int lane, int t, int s) {
  const unsigned int* p = B + ((t * 4 + s) * 32 + lane) * 8;
  v8h lo = *(const v8h*)p;        // v = 0..3
  v8h hi = *(const v8h*)(p + 4);  // v = 4..7
  return cat8(lo, hi);
}

// One wave: C[16x128] += A[16x128] * B[128x128], f16 in, f32 accum.
static __device__ __forceinline__ void gemm16x128(const _Float16* A, const unsigned int* B,
                                                  int lane, v8f acc[8]) {
#pragma unroll
  for (int s = 0; s < 4; ++s) {
    v16h a = load_afrag(A, lane, s);
#pragma unroll
    for (int t = 0; t < 8; ++t) {
      v16h b = load_bfrag(B, lane, t, s);
      acc[t] = __builtin_amdgcn_wmma_f32_16x16x32_f16(
          false, a, false, b, (short)0, acc[t], false, false);
    }
  }
}

// Per-row LayerNorm stats over the 128 columns of a wave's C tile.
// Lane holds cols {16t + lane&15}; reduce 8 accum tiles then xor-shuffle the
// 16-lane group (masks 1,2,4,8 stay inside each half-wave group).
static __device__ __forceinline__ void ln_stats(const v8f acc[8], float mu[8], float rs[8]) {
#pragma unroll
  for (int r = 0; r < 8; ++r) {
    float s1 = 0.f, s2 = 0.f;
#pragma unroll
    for (int t = 0; t < 8; ++t) { float x = acc[t][r]; s1 += x; s2 += x * x; }
#pragma unroll
    for (int d = 1; d < 16; d <<= 1) {
      s1 += __shfl_xor(s1, d, 32);
      s2 += __shfl_xor(s2, d, 32);
    }
    float m = s1 * (1.f / 128.f);
    float v = s2 * (1.f / 128.f) - m * m;
    mu[r] = m;
    rs[r] = rsqrtf(v + LN_EPS);
  }
}

static __device__ __forceinline__ void ln_apply_store(const v8f acc[8], const float* mu,
                                                      const float* rs, const float* g,
                                                      const float* b, _Float16* dst, int lane,
                                                      bool relu) {
  int c = lane & 15, gr = lane >> 4;
#pragma unroll
  for (int t = 0; t < 8; ++t) {
    int n = t * 16 + c;
    float gg = g[n], bb = b[n];
#pragma unroll
    for (int r = 0; r < 8; ++r) {
      float x = (acc[t][r] - mu[r]) * rs[r] * gg + bb;
      if (relu) x = fmaxf(x, 0.f);
      dst[(r + 8 * gr) * 128 + n] = (_Float16)x;
    }
  }
}

// ---------------------------------------------------------------------------
// Kernel 1: sWs = src@Ws, sWt = src@Wt, sVs = src@Vs  (tiny, VALU)
// ---------------------------------------------------------------------------
__global__ void proj_kernel(const float* __restrict__ src, const float* __restrict__ w_w1,
                            const float* __restrict__ v_w1, float* __restrict__ sWs,
                            float* __restrict__ sWt, float* __restrict__ sVs) {
  __shared__ float row[F];
  int r = blockIdx.x, m = blockIdx.y, t = threadIdx.x;
  row[t] = src[r * F + t];
  __syncthreads();
  const float* W = (m == 0) ? w_w1 : (m == 1) ? (w_w1 + 2 * F * F) : v_w1;
  float* O = (m == 0) ? sWs : (m == 1) ? sWt : sVs;
  float acc = 0.f;
#pragma unroll 8
  for (int k = 0; k < F; ++k) acc += row[k] * W[k * F + t];
  O[r * F + t] = acc;
}

// ---------------------------------------------------------------------------
// Kernel 2: fused pairwise attention (5 WMMA GEMMs per 8x16 pair tile)
// ---------------------------------------------------------------------------
__global__ void __launch_bounds__(256)
pair_kernel(const float* __restrict__ coords, const unsigned char* __restrict__ masks,
            const float* __restrict__ pe_w1, const float* __restrict__ pe_b1,
            const float* __restrict__ pe_w2, const float* __restrict__ pe_ln_g,
            const float* __restrict__ pe_ln_b, const float* __restrict__ w_w1,
            const float* __restrict__ w_ln_g, const float* __restrict__ w_ln_b,
            const float* __restrict__ w_w2, const float* __restrict__ w_b2,
            const float* __restrict__ v_w1, const float* __restrict__ v_ln_g,
            const float* __restrict__ v_ln_b, const float* __restrict__ v_w2,
            const float* __restrict__ v_b2, const float* __restrict__ sWs,
            const float* __restrict__ sWt, const float* __restrict__ sVs,
            float* __restrict__ agg) {
  extern __shared__ char smem[];
  unsigned int* Bu = (unsigned int*)smem;  // 5 * 8192 u32 = 160KB, frag-major f16

  const int tid = threadIdx.x;
  const int wave = tid >> 5, lane = tid & 31;
  const int i = blockIdx.x * IPB + wave;

  // ---- stage the 5 weight matrices as f16 B-fragments in LDS (once) ----
  {
    const float* Wsrc[5] = { pe_w2, w_w1 + F * F, v_w1 + F * F, w_w2, v_w2 };
    for (int idx = tid; idx < 5 * 8192; idx += 256) {
      int mat = idx >> 13;
      int r = idx & 8191;
      int v = r & 7, l = (r >> 3) & 31, s = (r >> 8) & 3, t = (r >> 10) & 7;
      int n = t * 16 + (l & 15);
      int k0 = s * 32 + (l >> 4) * 8 + (v >> 2) * 16 + 2 * (v & 3);  // B layout: table order, K 8g..8g+7 then 16+8g..16+8g+7
      const float* W = Wsrc[mat];
      v2h pk;
      pk.x = (_Float16)W[k0 * F + n];
      pk.y = (_Float16)W[(k0 + 1) * F + n];
      Bu[idx] = __builtin_bit_cast(unsigned int, pk);
    }
  }
  __syncthreads();

  const unsigned int* B_pe = Bu;
  const unsigned int* B_wp = Bu + 8192;
  const unsigned int* B_vp = Bu + 16384;
  const unsigned int* B_w2 = Bu + 24576;
  const unsigned int* B_v2 = Bu + 32768;

  // per-wave A staging buffers (16x128 f16 each): A0 hidden/aw, A1 pe, A2 av
  _Float16* A0 = (_Float16*)(smem + 163840 + wave * 12288);
  _Float16* A1 = A0 + 2048;
  _Float16* A2 = A0 + 4096;

  const float cx = coords[i * 2 + 0], cy = coords[i * 2 + 1];
  const int c = lane & 15, gr = lane >> 4;

  float aggacc[8];
#pragma unroll
  for (int t = 0; t < 8; ++t) aggacc[t] = 0.f;

  const v8f vzero = {};

  for (int j0 = 0; j0 < N; j0 += 16) {
    // ---- hidden = relu(rel @ pe_w1 + pe_b1): 2 lanes per row, 64 feats each
    {
      int m = lane >> 1;
      int j = j0 + m;
      float rx = cx - coords[j * 2 + 0];
      float ry = cy - coords[j * 2 + 1];
      int f0 = (lane & 1) * 64;
#pragma unroll 8
      for (int f = f0; f < f0 + 64; ++f) {
        float h = rx * pe_w1[f] + ry * pe_w1[F + f] + pe_b1[f];
        A0[m * 128 + f] = (_Float16)fmaxf(h, 0.f);
      }
    }
    asm volatile("s_wait_dscnt 0x0" ::: "memory");

    // ---- GEMM1: pe_raw = hidden @ pe_w2 ; pe = LN(pe_raw) -> A1
    {
      v8f acc[8];
#pragma unroll
      for (int t = 0; t < 8; ++t) acc[t] = vzero;
      gemm16x128(A0, B_pe, lane, acc);
      float mu[8], rs[8];
      ln_stats(acc, mu, rs);
      ln_apply_store(acc, mu, rs, pe_ln_g, pe_ln_b, A1, lane, false);
    }
    asm volatile("s_wait_dscnt 0x0" ::: "memory");

    // ---- GEMM2: pre_w = pe@Wp + sWs[j] + sWt[i]; aw = relu(LN(pre_w)) -> A0
    {
      v8f acc[8];
#pragma unroll
      for (int t = 0; t < 8; ++t) acc[t] = vzero;
      gemm16x128(A1, B_wp, lane, acc);
#pragma unroll
      for (int t = 0; t < 8; ++t) {
        int n = t * 16 + c;
        float swt = sWt[i * F + n];
#pragma unroll
        for (int r = 0; r < 8; ++r) {
          int j = j0 + r + 8 * gr;
          acc[t][r] += sWs[j * F + n] + swt;
        }
      }
      float mu[8], rs[8];
      ln_stats(acc, mu, rs);
      ln_apply_store(acc, mu, rs, w_ln_g, w_ln_b, A0, lane, true);
    }
    asm volatile("s_wait_dscnt 0x0" ::: "memory");

    // ---- GEMM3: pre_v = pe@Vp + sVs[j]; av = relu(LN(pre_v)) -> A2
    {
      v8f acc[8];
#pragma unroll
      for (int t = 0; t < 8; ++t) acc[t] = vzero;
      gemm16x128(A1, B_vp, lane, acc);
#pragma unroll
      for (int t = 0; t < 8; ++t) {
        int n = t * 16 + c;
#pragma unroll
        for (int r = 0; r < 8; ++r) {
          int j = j0 + r + 8 * gr;
          acc[t][r] += sVs[j * F + n];
        }
      }
      float mu[8], rs[8];
      ln_stats(acc, mu, rs);
      ln_apply_store(acc, mu, rs, v_ln_g, v_ln_b, A2, lane, true);
    }
    asm volatile("s_wait_dscnt 0x0" ::: "memory");

    // ---- GEMM4/5: weight = sigmoid(aw@w_w2 + w_b2), value = av@v_w2 + v_b2
    {
      v8f accW[8], accV[8];
#pragma unroll
      for (int t = 0; t < 8; ++t) { accW[t] = vzero; accV[t] = vzero; }
      gemm16x128(A0, B_w2, lane, accW);
      gemm16x128(A2, B_v2, lane, accV);

      // masked aggregation: sum over the 16 j-rows of this tile
#pragma unroll
      for (int t = 0; t < 8; ++t) {
        int n = t * 16 + c;
        float wb = w_b2[n], vb = v_b2[n];
        float partial = 0.f;
#pragma unroll
        for (int r = 0; r < 8; ++r) {
          int j = j0 + r + 8 * gr;
          float wgt = 1.f / (1.f + __expf(-(accW[t][r] + wb)));
          float val = accV[t][r] + vb;
          float mk = (float)masks[i * N + j];
          partial += mk * wgt * val;
        }
        partial += __shfl_xor(partial, 16, 32);  // combine both row groups
        aggacc[t] += partial;
      }
    }
  }

  // one i per wave: lanes 0..15 each write 8 columns
  if (lane < 16) {
#pragma unroll
    for (int t = 0; t < 8; ++t) agg[i * F + t * 16 + lane] = aggacc[t];
  }
}

// ---------------------------------------------------------------------------
// Kernel 3: out = relu(LN(relu(LN(agg)) @ out_w) + src)  (tiny)
// ---------------------------------------------------------------------------
__global__ void final_kernel(const float* __restrict__ agg, const float* __restrict__ src,
                             const float* __restrict__ norm_g, const float* __restrict__ norm_b,
                             const float* __restrict__ out_w, const float* __restrict__ out_ln_g,
                             const float* __restrict__ out_ln_b, float* __restrict__ out) {
  __shared__ float sh[F];
  __shared__ float red[8];
  int r = blockIdx.x, t = threadIdx.x, lane = t & 31, w = t >> 5;

  float x = agg[r * F + t];
  float s1 = x, s2 = x * x;
#pragma unroll
  for (int d = 1; d < 32; d <<= 1) { s1 += __shfl_xor(s1, d, 32); s2 += __shfl_xor(s2, d, 32); }
  if (lane == 0) { red[w] = s1; red[4 + w] = s2; }
  __syncthreads();
  float mu = (red[0] + red[1] + red[2] + red[3]) * (1.f / 128.f);
  float var = (red[4] + red[5] + red[6] + red[7]) * (1.f / 128.f) - mu * mu;
  float rs = rsqrtf(var + LN_EPS);
  float h = fmaxf((x - mu) * rs * norm_g[t] + norm_b[t], 0.f);
  sh[t] = h;
  __syncthreads();

  float o = 0.f;
#pragma unroll 8
  for (int k = 0; k < F; ++k) o += sh[k] * out_w[k * F + t];

  s1 = o; s2 = o * o;
#pragma unroll
  for (int d = 1; d < 32; d <<= 1) { s1 += __shfl_xor(s1, d, 32); s2 += __shfl_xor(s2, d, 32); }
  __syncthreads();
  if (lane == 0) { red[w] = s1; red[4 + w] = s2; }
  __syncthreads();
  float mu2 = (red[0] + red[1] + red[2] + red[3]) * (1.f / 128.f);
  float var2 = (red[4] + red[5] + red[6] + red[7]) * (1.f / 128.f) - mu2 * mu2;
  float rs2 = rsqrtf(var2 + LN_EPS);
  float y = (o - mu2) * rs2 * out_ln_g[t] + out_ln_b[t];
  out[r * F + t] = fmaxf(y + src[r * F + t], 0.f);
}

// ---------------------------------------------------------------------------
extern "C" void kernel_launch(void* const* d_in, const int* in_sizes, int n_in,
                              void* d_out, int out_size, void* d_ws, size_t ws_size,
                              hipStream_t stream) {
  const float* src      = (const float*)d_in[0];
  const float* coords   = (const float*)d_in[1];
  const unsigned char* masks = (const unsigned char*)d_in[2];
  const float* pe_w1    = (const float*)d_in[3];
  const float* pe_b1    = (const float*)d_in[4];
  const float* pe_w2    = (const float*)d_in[5];
  const float* pe_ln_g  = (const float*)d_in[6];
  const float* pe_ln_b  = (const float*)d_in[7];
  const float* w_w1     = (const float*)d_in[8];
  const float* w_ln_g   = (const float*)d_in[9];
  const float* w_ln_b   = (const float*)d_in[10];
  const float* w_w2     = (const float*)d_in[11];
  const float* w_b2     = (const float*)d_in[12];
  const float* v_w1     = (const float*)d_in[13];
  const float* v_ln_g   = (const float*)d_in[14];
  const float* v_ln_b   = (const float*)d_in[15];
  const float* v_w2     = (const float*)d_in[16];
  const float* v_b2     = (const float*)d_in[17];
  const float* norm_g   = (const float*)d_in[18];
  const float* norm_b   = (const float*)d_in[19];
  const float* out_w    = (const float*)d_in[20];
  const float* out_ln_g = (const float*)d_in[21];
  const float* out_ln_b = (const float*)d_in[22];

  float* ws  = (float*)d_ws;
  float* sWs = ws;
  float* sWt = ws + N * F;
  float* sVs = ws + 2 * N * F;
  float* agg = ws + 3 * N * F;
  float* out = (float*)d_out;

  proj_kernel<<<dim3(N, 3), F, 0, stream>>>(src, w_w1, v_w1, sWs, sWt, sVs);

  size_t lds = 163840 + IPB * 12288;  // 160KB B-frags + 8 waves * 12KB A-bufs = 256KB
  pair_kernel<<<N / IPB, 256, lds, stream>>>(
      coords, masks, pe_w1, pe_b1, pe_w2, pe_ln_g, pe_ln_b, w_w1, w_ln_g, w_ln_b,
      w_w2, w_b2, v_w1, v_ln_g, v_ln_b, v_w2, v_b2, sWs, sWt, sVs, agg);

  final_kernel<<<N, F, 0, stream>>>(agg, src, norm_g, norm_b, out_w, out_ln_g,
                                    out_ln_b, out);
}
